// DualTokenQFormer_74998718922896
// MI455X (gfx1250) — matrix operand order repaired and checked by reading, not compiled, not run on hardware
//
#include <hip/hip_runtime.h>
#include <math.h>

// ---------------- problem constants ----------------
#define NTOK   32768
#define DMODEL 1024
#define NHEAD  16
#define HDIM   64
#define NLAYER 2
#define DFFN   4096
#define KTOP   8192   // 0.25 * N

// ---------------- CDNA5 WMMA vector types ----------------
typedef __attribute__((ext_vector_type(16))) __bf16 v16bf;
typedef __attribute__((ext_vector_type(8)))  __bf16 v8bf;
typedef __attribute__((ext_vector_type(8)))  float  v8f;

__device__ __forceinline__ unsigned short f2bf(float f) {
    unsigned u = __float_as_uint(f);
    unsigned r = u + 0x7FFFu + ((u >> 16) & 1u);   // round-to-nearest-even
    return (unsigned short)(r >> 16);
}
__device__ __forceinline__ float bf2f(unsigned short h) {
    return __uint_as_float(((unsigned)h) << 16);
}

// ---------------- CDNA5 async global->LDS copy (ASYNCcnt-tracked DMA path) ----
// vdst VGPR carries the per-lane LDS byte address; vaddr the 64-bit global addr.
__device__ __forceinline__ void async_g2l_b128(unsigned lds_off, const void* gptr) {
    unsigned long long ga = (unsigned long long)gptr;
    asm volatile("global_load_async_to_lds_b128 %0, %1, off"
                 :: "v"(lds_off), "v"(ga)
                 : "memory");
}
__device__ __forceinline__ void wait_async0() {
#if __has_builtin(__builtin_amdgcn_s_wait_asynccnt)
    __builtin_amdgcn_s_wait_asynccnt(0);
#else
    asm volatile("s_wait_asynccnt 0" ::: "memory");
#endif
}

// ---------------- stage X as bf16 (halve HBM traffic for all streaming passes) ----
__global__ __launch_bounds__(256) void k_xconvert(const float* __restrict__ X,
                                                  unsigned short* __restrict__ Xb) {
    size_t i = ((size_t)blockIdx.x * 256 + threadIdx.x) * 4;
    float4 v = *(const float4*)(X + i);
    Xb[i + 0] = f2bf(v.x); Xb[i + 1] = f2bf(v.y);
    Xb[i + 2] = f2bf(v.z); Xb[i + 3] = f2bf(v.w);
}

// ---------------- saliency score: ||x_j||_2 (block per token) ----------------
__global__ __launch_bounds__(256) void k_saliency(const float* __restrict__ X,
                                                  float* __restrict__ score) {
    __shared__ float red[256];
    const float* xr = X + (size_t)blockIdx.x * DMODEL;
    float s = 0.f;
    for (int i = threadIdx.x; i < DMODEL; i += 256) { float v = xr[i]; s += v * v; }
    red[threadIdx.x] = s; __syncthreads();
    for (int o = 128; o; o >>= 1) { if (threadIdx.x < o) red[threadIdx.x] += red[threadIdx.x + o]; __syncthreads(); }
    if (!threadIdx.x) score[blockIdx.x] = sqrtf(red[0]);
}

// ---------------- per-dim mean/max partials over token chunks ----------------
__global__ __launch_bounds__(256) void k_meanmax_part(const float* __restrict__ X,
                                                      float* __restrict__ mpart,
                                                      float* __restrict__ xpart) {
    int d = blockIdx.x * 256 + threadIdx.x;          // 0..1023  (grid.x = 4)
    int c = blockIdx.y;                              // 32 chunks of 1024 tokens
    size_t j0 = (size_t)c * 1024;
    float s = 0.f, m = -1e30f;
    for (int j = 0; j < 1024; ++j) {
        float v = X[(j0 + j) * DMODEL + d];
        s += v; m = fmaxf(m, v);
    }
    mpart[c * DMODEL + d] = s;
    xpart[c * DMODEL + d] = m;
}

// ---------------- exact top-k by rank counting (LDS-tiled scores) ----------------
__global__ __launch_bounds__(256) void k_topk(const float* __restrict__ score,
                                              int* __restrict__ topk, int n, int k) {
    __shared__ float sc[4096];
    int i = blockIdx.x * 256 + threadIdx.x;
    float s = score[i];
    int rank = 0;
    for (int c0 = 0; c0 < n; c0 += 4096) {
        for (int e = threadIdx.x; e < 4096; e += 256) sc[e] = score[c0 + e];
        __syncthreads();
        for (int e = 0; e < 4096; ++e) {
            float o = sc[e];
            rank += (o > s) || (o == s && (c0 + e) < i);
        }
        __syncthreads();
    }
    if (rank < k) topk[rank] = i;    // permutation of the exact top-k set
}

// ---------------- query init: [mean; max] + learned embeds ----------------
__global__ __launch_bounds__(256) void k_qinit(const float* __restrict__ mpart,
                                               const float* __restrict__ xpart,
                                               const float* __restrict__ qb,
                                               const float* __restrict__ re,
                                               const float* __restrict__ te,
                                               float* __restrict__ q) {
    int i = blockIdx.x * 256 + threadIdx.x;          // 2048
    int r = i >> 10, d = i & 1023;
    float val;
    if (r == 0) { float s = 0.f; for (int c = 0; c < 32; ++c) s += mpart[c * DMODEL + d]; val = s / (float)NTOK; }
    else        { float m = -1e30f; for (int c = 0; c < 32; ++c) m = fmaxf(m, xpart[c * DMODEL + d]); val = m; }
    q[i] = val + qb[i] + re[i] + te[d];
}

// ---------------- LayerNorm over rows of 1024 ----------------
__global__ __launch_bounds__(256) void k_lnrow(const float* __restrict__ x,
                                               const float* __restrict__ g,
                                               const float* __restrict__ b,
                                               float* __restrict__ y) {
    __shared__ float red[256];
    int row = blockIdx.x, tid = threadIdx.x;
    const float* xr = x + (size_t)row * DMODEL;
    float s = 0.f;
    for (int i = tid; i < DMODEL; i += 256) s += xr[i];
    red[tid] = s; __syncthreads();
    for (int o = 128; o; o >>= 1) { if (tid < o) red[tid] += red[tid + o]; __syncthreads(); }
    float mean = red[0] * (1.f / DMODEL); __syncthreads();
    float v = 0.f;
    for (int i = tid; i < DMODEL; i += 256) { float d = xr[i] - mean; v += d * d; }
    red[tid] = v; __syncthreads();
    for (int o = 128; o; o >>= 1) { if (tid < o) red[tid] += red[tid + o]; __syncthreads(); }
    float rstd = rsqrtf(red[0] * (1.f / DMODEL) + 1e-5f); __syncthreads();
    for (int i = tid; i < DMODEL; i += 256)
        y[(size_t)row * DMODEL + i] = (xr[i] - mean) * rstd * g[i] + b[i];
}

// ---------------- generic small GEMV: Y[r,c] = act(sum_i X[r,i]*W[i,wcol0+c] + B[c]) + resid
// op: 0 = none, 1 = exact GELU
__global__ __launch_bounds__(256) void k_gemv(const float* __restrict__ X, int xstride,
                                              const float* __restrict__ W, int wld, int wcol0,
                                              const float* __restrict__ B,
                                              const float* __restrict__ resid, int rstride,
                                              float* __restrict__ Y, int ystride,
                                              int K, int op) {
    __shared__ float xs[4096];
    int row = blockIdx.y;
    const float* xr = X + (size_t)row * xstride;
    for (int i = threadIdx.x; i < K; i += 256) xs[i] = xr[i];
    __syncthreads();
    int c = blockIdx.x * 256 + threadIdx.x;
    float acc = B ? B[c] : 0.f;
    const float* wp = W + (size_t)wcol0 + c;
    for (int i = 0; i < K; ++i) acc += xs[i] * wp[(size_t)i * wld];
    if (op == 1) acc = 0.5f * acc * (1.f + erff(acc * 0.70710678118654752f));
    if (resid) acc += resid[(size_t)row * rstride + c];
    Y[(size_t)row * ystride + c] = acc;
}

// ---------------- 2x2 self-attention (tiny) ----------------
__global__ __launch_bounds__(256) void k_selfattn2(const float* __restrict__ qkv,
                                                   float* __restrict__ out) {
    __shared__ float att[NHEAD][4];
    int tid = threadIdx.x;
    if (tid < 64) {
        int h = tid >> 2, qi = (tid >> 1) & 1, ki = tid & 1;
        float s = 0.f;
        for (int c = 0; c < HDIM; ++c)
            s += qkv[qi * 3072 + h * HDIM + c] * qkv[ki * 3072 + 1024 + h * HDIM + c];
        att[h][qi * 2 + ki] = s * 0.125f;            // 1/sqrt(64)
    }
    __syncthreads();
    if (tid < 32) {
        int h = tid >> 1, qi = tid & 1;
        float a = att[h][qi * 2], b = att[h][qi * 2 + 1];
        float m = fmaxf(a, b);
        float ea = __expf(a - m), eb = __expf(b - m);
        float inv = 1.f / (ea + eb);
        att[h][qi * 2] = ea * inv; att[h][qi * 2 + 1] = eb * inv;
    }
    __syncthreads();
    for (int i = tid; i < 2 * DMODEL; i += 256) {
        int r = i >> 10, d = i & 1023, h = d >> 6;
        out[i] = att[h][r * 2 + 0] * qkv[0 * 3072 + 2048 + d]
               + att[h][r * 2 + 1] * qkv[1 * 3072 + 2048 + d];
    }
}

// ---------------- M build: M[h,d] = (1/8) * sum_c Wk[d, h*64+c] * qp[h*64+c]  (bf16 out)
__global__ __launch_bounds__(256) void k_mbuild(const float* __restrict__ qp,
                                                const float* __restrict__ win,
                                                unsigned short* __restrict__ Mb) {
    __shared__ float qs[DMODEL];
    for (int i = threadIdx.x; i < DMODEL; i += 256) qs[i] = qp[i];
    __syncthreads();
    int t = blockIdx.x * 256 + threadIdx.x;          // 16384
    int h = t >> 10, d = t & 1023;
    const float* wc = win + (size_t)d * 3072 + 1024 + h * HDIM;
    float acc = 0.f;
    for (int c = 0; c < HDIM; ++c) acc += wc[c] * qs[h * HDIM + c];
    Mb[t] = f2bf(acc * 0.125f);
}

// ---------------- WMMA scores: St[h, j] = (X_kv @ M^T)[j, h]  ----------------
// One wave computes a 16-token x 16-head tile, K=1024 in steps of 32 (bf16 WMMA).
// B panel (M, 16x1024 bf16 = 32 KB) is reused by all 8 waves x 32 k-steps of the
// block -> stage it once via CDNA5 async global->LDS DMA, feed WMMA B from LDS.
// A k order chosen here (contiguous, same as B): lane<16 holds K {0..15}, lane>=16 {16..31}.
// B k order chosen here (same as A): lane&15 = column(head), K = (lane>>4)*16 + e.
__global__ __launch_bounds__(256) void k_scores_wmma(const unsigned short* __restrict__ Xb,
                                                     const int* __restrict__ idx,
                                                     const unsigned short* __restrict__ Mb,
                                                     float* __restrict__ St, int cnt) {
    __shared__ unsigned short bsm[NHEAD * DMODEL];   // 32 KB B panel in LDS
    {
        unsigned lds0 = (unsigned)(unsigned long long)(&bsm[0]); // low 32b = LDS addr
        int t = threadIdx.x;
#pragma unroll
        for (int c = 0; c < 8; ++c) {                 // 256 thr * 8 * 16B = 32 KB
            unsigned o = (unsigned)((t + c * 256) * 16);
            async_g2l_b128(lds0 + o, (const char*)Mb + o);
        }
        wait_async0();
        __syncthreads();
    }

    int lane = threadIdx.x & 31;
    int wv   = (int)((blockIdx.x * blockDim.x + threadIdx.x) >> 5);
    int tokBase = wv << 4;
    int half = lane >> 4, l15 = lane & 15;
    int row  = tokBase + l15;
    int grow = idx ? idx[row] : row;
    const unsigned short* xr = Xb + (size_t)grow * DMODEL;
    v8f acc = {};
#pragma unroll 4
    for (int kk = 0; kk < DMODEL; kk += 32) {
        v16bf a = *(const v16bf*)(xr + kk + half * 16);
        v16bf b = *(const v16bf*)(&bsm[l15 * DMODEL + kk + half * 16]);   // ds_load
        acc = __builtin_amdgcn_wmma_f32_16x16x32_bf16(false, a, false, b,
                                                      (short)0, acc, false, false);
    }
    // C/D: VGPR r -> token m = r + 8*half, lane&15 -> head
    float* so = St + (size_t)l15 * cnt + tokBase;
#pragma unroll
    for (int r = 0; r < 8; ++r) so[r + 8 * half] = acc[r];
}

// ---------------- per-head softmax stats (max + sumexp) ----------------
__global__ __launch_bounds__(256) void k_softmax_stats(const float* __restrict__ St,
                                                       float* __restrict__ hmax,
                                                       float* __restrict__ hsum, int cnt) {
    __shared__ float red[256];
    int h = blockIdx.x, tid = threadIdx.x;
    const float* s = St + (size_t)h * cnt;
    float m = -1e30f;
    for (int i = tid; i < cnt; i += 256) m = fmaxf(m, s[i]);
    red[tid] = m; __syncthreads();
    for (int o = 128; o; o >>= 1) { if (tid < o) red[tid] = fmaxf(red[tid], red[tid + o]); __syncthreads(); }
    float mx = red[0]; __syncthreads();
    float sum = 0.f;
    for (int i = tid; i < cnt; i += 256) sum += __expf(s[i] - mx);
    red[tid] = sum; __syncthreads();
    for (int o = 128; o; o >>= 1) { if (tid < o) red[tid] += red[tid + o]; __syncthreads(); }
    if (!tid) { hmax[h] = mx; hsum[h] = red[0]; }
}

__global__ __launch_bounds__(256) void k_exp_inplace(float* __restrict__ St,
                                                     const float* __restrict__ hmax, int cnt) {
    int i = blockIdx.x * 256 + threadIdx.x;
    int h = i / cnt;
    St[i] = __expf(St[i] - hmax[h]);
}

// ---------------- rank-16 weighted sum: Wpart[b,h,d] = sum_j p[h,j] * x_j[d] ------
// Each thread owns 4 dims, accumulates 16 heads in registers; the probability
// tile for each 64-token chunk is staged into LDS via async global->LDS DMA.
__global__ __launch_bounds__(256) void k_weighted_partial(const unsigned short* __restrict__ Xb,
                                                          const int* __restrict__ idx,
                                                          const float* __restrict__ Pt,
                                                          float* __restrict__ Wpart, int cnt) {
    __shared__ float pch[NHEAD * 64];                // 4 KB probability tile
    int tid = threadIdx.x;
    int j0 = blockIdx.x * 256;
    unsigned plds = (unsigned)(unsigned long long)(&pch[0]);
    float acc[NHEAD][4];
#pragma unroll
    for (int h = 0; h < NHEAD; ++h) { acc[h][0] = acc[h][1] = acc[h][2] = acc[h][3] = 0.f; }
    for (int jc = j0; jc < j0 + 256; jc += 64) {
        // stage p[h, jc..jc+63]: thread -> 4 consecutive floats of one head row
        {
            int h  = tid >> 4;                        // 16 b128-groups per 64-wide row
            int jj = (tid & 15) * 4;
            async_g2l_b128(plds + (unsigned)tid * 16,
                           Pt + (size_t)h * cnt + jc + jj);
            wait_async0();
        }
        __syncthreads();
        for (int jj = 0; jj < 64; ++jj) {
            int j = jc + jj;
            int g = idx ? idx[j] : j;
            const unsigned short* xr = Xb + (size_t)g * DMODEL + tid * 4;
            unsigned long long bits = *(const unsigned long long*)xr;   // 4 bf16
            float x0 = bf2f((unsigned short)bits);
            float x1 = bf2f((unsigned short)(bits >> 16));
            float x2 = bf2f((unsigned short)(bits >> 32));
            float x3 = bf2f((unsigned short)(bits >> 48));
#pragma unroll
            for (int h = 0; h < NHEAD; ++h) {
                float p = pch[h * 64 + jj];
                acc[h][0] += p * x0; acc[h][1] += p * x1;
                acc[h][2] += p * x2; acc[h][3] += p * x3;
            }
        }
        __syncthreads();
    }
    float* wp = Wpart + (size_t)blockIdx.x * NHEAD * DMODEL;
#pragma unroll
    for (int h = 0; h < NHEAD; ++h)
#pragma unroll
        for (int i = 0; i < 4; ++i)
            wp[h * DMODEL + tid * 4 + i] = acc[h][i];
}

__global__ __launch_bounds__(256) void k_wreduce(const float* __restrict__ Wpart,
                                                 float* __restrict__ Wsum, int nblk) {
    int i = blockIdx.x * 256 + threadIdx.x;          // 16384
    float s = 0.f;
    for (int b = 0; b < nblk; ++b) s += Wpart[(size_t)b * NHEAD * DMODEL + i];
    Wsum[i] = s;
}

// ---------------- V projection: vout[h*64+c] = Wsum[h]·Wv[:,h*64+c]/hsum[h] + bv --
__global__ __launch_bounds__(256) void k_vproj(const float* __restrict__ Wsum,
                                               const float* __restrict__ hsum,
                                               const float* __restrict__ win,
                                               const float* __restrict__ bin,
                                               float* __restrict__ vout) {
    int cg = blockIdx.x * 256 + threadIdx.x;         // 1024
    int h = cg >> 6;
    const float* wr = Wsum + (size_t)h * DMODEL;
    const float* wp = win + 2048 + cg;               // column 2D+cg, row stride 3072
    float acc = 0.f;
    for (int d = 0; d < DMODEL; ++d) acc += wr[d] * wp[(size_t)d * 3072];
    vout[cg] = acc / hsum[h] + bin[2048 + cg];
}

// =====================================================================
extern "C" void kernel_launch(void* const* d_in, const int* in_sizes, int n_in,
                              void* d_out, int out_size, void* d_ws, size_t ws_size,
                              hipStream_t stream) {
    const float* frame  = (const float*)d_in[0];
    const float* qbase  = (const float*)d_in[1];
    const float* rolee  = (const float*)d_in[2];
    const float* temb   = (const float*)d_in[3];
    const float* ln1g   = (const float*)d_in[4];
    const float* ln1b   = (const float*)d_in[5];
    const float* ln2g   = (const float*)d_in[6];
    const float* ln2b   = (const float*)d_in[7];
    const float* ln3g   = (const float*)d_in[8];
    const float* ln3b   = (const float*)d_in[9];
    const float* sa_win = (const float*)d_in[10];
    const float* sa_bin = (const float*)d_in[11];
    const float* sa_wout= (const float*)d_in[12];
    const float* sa_bout= (const float*)d_in[13];
    const float* cg_win = (const float*)d_in[14];
    const float* cg_bin = (const float*)d_in[15];
    const float* cg_wout= (const float*)d_in[16];
    const float* cg_bout= (const float*)d_in[17];
    const float* cs_win = (const float*)d_in[18];
    const float* cs_bin = (const float*)d_in[19];
    const float* cs_wout= (const float*)d_in[20];
    const float* cs_bout= (const float*)d_in[21];
    const float* f1w    = (const float*)d_in[22];
    const float* f1b    = (const float*)d_in[23];
    const float* f2w    = (const float*)d_in[24];
    const float* f2b    = (const float*)d_in[25];
    const float* outg   = (const float*)d_in[26];
    const float* outb   = (const float*)d_in[27];
    (void)in_sizes; (void)n_in; (void)out_size; (void)ws_size;

    // ---- workspace carve (256B aligned) ----
    char* w = (char*)d_ws;
    size_t off = 0;
    auto carve = [&](size_t bytes) { void* p = w + off; off = (off + bytes + 255) & ~(size_t)255; return p; };
    unsigned short* Xb   = (unsigned short*)carve((size_t)NTOK * DMODEL * 2);   // 64 MB
    float* sal   = (float*)carve(NTOK * 4);
    int*   topk  = (int*)  carve(KTOP * 4);
    float* mpart = (float*)carve(32 * DMODEL * 4);
    float* xpart = (float*)carve(32 * DMODEL * 4);
    float* q     = (float*)carve(2 * DMODEL * 4);
    float* qln   = (float*)carve(2 * DMODEL * 4);
    float* qkv   = (float*)carve(2 * 3 * DMODEL * 4);
    float* attno = (float*)carve(2 * DMODEL * 4);
    float* hbuf  = (float*)carve(2 * DFFN * 4);
    float* qp    = (float*)carve(DMODEL * 4);
    unsigned short* Mb = (unsigned short*)carve(NHEAD * DMODEL * 2);
    float* St    = (float*)carve((size_t)NHEAD * NTOK * 4);                     // 2 MB
    float* hmax  = (float*)carve(NHEAD * 4);
    float* hsum  = (float*)carve(NHEAD * 4);
    float* Wpart = (float*)carve((size_t)(NTOK / 256) * NHEAD * DMODEL * 4);    // 8 MB
    float* Wsum  = (float*)carve(NHEAD * DMODEL * 4);
    float* voutb = (float*)carve(DMODEL * 4);

    dim3 B(256);

    // ---- setup ----
    k_xconvert   <<<(NTOK * DMODEL) / 1024, B, 0, stream>>>(frame, Xb);
    k_saliency   <<<NTOK, B, 0, stream>>>(frame, sal);
    k_meanmax_part<<<dim3(4, 32), B, 0, stream>>>(frame, mpart, xpart);
    k_topk       <<<NTOK / 256, B, 0, stream>>>(sal, topk, NTOK, KTOP);
    k_qinit      <<<8, B, 0, stream>>>(mpart, xpart, qbase, rolee, temb, q);

    for (int li = 0; li < NLAYER; ++li) {
        const float* saw  = sa_win  + (size_t)li * DMODEL * 3 * DMODEL;
        const float* sab  = sa_bin  + (size_t)li * 3 * DMODEL;
        const float* sawo = sa_wout + (size_t)li * DMODEL * DMODEL;
        const float* sabo = sa_bout + (size_t)li * DMODEL;

        // --- self-attention block ---
        k_lnrow<<<2, B, 0, stream>>>(q, ln1g + li * DMODEL, ln1b + li * DMODEL, qln);
        k_gemv <<<dim3(12, 2), B, 0, stream>>>(qln, DMODEL, saw, 3072, 0, sab,
                                               nullptr, 0, qkv, 3072, DMODEL, 0);
        k_selfattn2<<<1, B, 0, stream>>>(qkv, attno);
        k_gemv <<<dim3(4, 2), B, 0, stream>>>(attno, DMODEL, sawo, DMODEL, 0, sabo,
                                              q, DMODEL, q, DMODEL, DMODEL, 0);

        // --- cross-attention: branch 0 = global (q row 0), branch 1 = salient (q row 1) ---
        k_lnrow<<<2, B, 0, stream>>>(q, ln2g + li * DMODEL, ln2b + li * DMODEL, qln);
        for (int br = 0; br < 2; ++br) {
            int row = br;
            int cnt = br ? KTOP : NTOK;
            const int*  gidx  = br ? topk : nullptr;
            const float* winl = (br ? cs_win  : cg_win ) + (size_t)li * DMODEL * 3 * DMODEL;
            const float* binl = (br ? cs_bin  : cg_bin ) + (size_t)li * 3 * DMODEL;
            const float* wol  = (br ? cs_wout : cg_wout) + (size_t)li * DMODEL * DMODEL;
            const float* bol  = (br ? cs_bout : cg_bout) + (size_t)li * DMODEL;

            // q projection for this single query row
            k_gemv<<<dim3(4, 1), B, 0, stream>>>(qln + (size_t)row * DMODEL, DMODEL,
                                                 winl, 3072, 0, binl,
                                                 nullptr, 0, qp, DMODEL, DMODEL, 0);
            // fold Wk into M (k-bias cancels in softmax; 1/sqrt(hd) folded here)
            k_mbuild<<<64, B, 0, stream>>>(qp, winl, Mb);
            // WMMA skinny GEMM: scores for all kv tokens, head-major
            k_scores_wmma<<<cnt / 128, B, 0, stream>>>(Xb, gidx, Mb, St, cnt);
            k_softmax_stats<<<NHEAD, B, 0, stream>>>(St, hmax, hsum, cnt);
            k_exp_inplace<<<NHEAD * cnt / 256, B, 0, stream>>>(St, hmax, cnt);
            // rank-16 weighted token sum (bandwidth-bound bf16 stream)
            k_weighted_partial<<<cnt / 256, B, 0, stream>>>(Xb, gidx, St, Wpart, cnt);
            k_wreduce<<<64, B, 0, stream>>>(Wpart, Wsum, cnt / 256);
            // fold Wv (v-bias adds since sum(p)=1), then out-proj + residual
            k_vproj<<<4, B, 0, stream>>>(Wsum, hsum, winl, binl, voutb);
            k_gemv<<<dim3(4, 1), B, 0, stream>>>(voutb, DMODEL, wol, DMODEL, 0, bol,
                                                 q + (size_t)row * DMODEL, DMODEL,
                                                 q + (size_t)row * DMODEL, DMODEL, DMODEL, 0);
        }

        // --- FFN ---
        k_lnrow<<<2, B, 0, stream>>>(q, ln3g + li * DMODEL, ln3b + li * DMODEL, qln);
        k_gemv<<<dim3(16, 2), B, 0, stream>>>(qln, DMODEL,
                                              f1w + (size_t)li * DMODEL * DFFN, DFFN, 0,
                                              f1b + (size_t)li * DFFN,
                                              nullptr, 0, hbuf, DFFN, DMODEL, 1 /*gelu*/);
        k_gemv<<<dim3(4, 2), B, 0, stream>>>(hbuf, DFFN,
                                             f2w + (size_t)li * DFFN * DMODEL, DMODEL, 0,
                                             f2b + (size_t)li * DMODEL,
                                             q, DMODEL, q, DMODEL, DFFN, 0);
    }

    // final LN -> output (2 x 1024 fp32)
    k_lnrow<<<2, B, 0, stream>>>(q, outg, outb, (float*)d_out);
}
